// DeepseekV4CSACompressor_89137751261355
// MI455X (gfx1250) — matrix operand order repaired and checked by reading, not compiled, not run on hardware
//
#include <hip/hip_runtime.h>
#include <hip/hip_bf16.h>
#include <math.h>

// ---------- types ----------
typedef __attribute__((ext_vector_type(16))) __bf16 v16bf;
typedef __attribute__((ext_vector_type(8)))  __bf16 v8bf;
typedef __attribute__((ext_vector_type(2)))  __bf16 bf2;
typedef __attribute__((ext_vector_type(8)))  float  v8f;
typedef __attribute__((ext_vector_type(4)))  float  f4v;
typedef __attribute__((ext_vector_type(2)))  float  f2v;

#define H_DIM   4096
#define N_OUT   512        // 2*HEAD_DIM kv ++ 2*HEAD_DIM gate
#define M_TOT   32768      // B*S
#define TILE_M  128
#define TILE_N  128
#define KSTEP   64
#define ASTRIDE 72         // 64 halves + 8 pad (144 B rows, 16B aligned)
#define BSTRIDE 72

__device__ __forceinline__ uint32_t pk2bf(float x, float y) {
#if __has_builtin(__builtin_amdgcn_cvt_pk_bf16_f32)
  bf2 r = __builtin_amdgcn_cvt_pk_bf16_f32(x, y);   // single v_cvt_pk_bf16_f32
#else
  f2v f = {x, y};
  bf2 r = __builtin_convertvector(f, bf2);
#endif
  union { bf2 b; uint32_t u; } o; o.b = r; return o.u;
}

// ---------- kernel 0: transpose + convert weights to bf16 ----------
// wT[n][k] for n in [0,512): n<256 -> w_kv col n ; n>=256 -> w_gate col n-256
__global__ __launch_bounds__(256) void prep_weights(
    const float* __restrict__ wkv, const float* __restrict__ wg,
    __bf16* __restrict__ wT)
{
  int idx = blockIdx.x * 256 + threadIdx.x;     // 512*4096 = 2M
  if (idx >= N_OUT * H_DIM) return;
  int n = idx >> 12;
  int k = idx & (H_DIM - 1);
  float v = (n < 256) ? wkv[(size_t)k * 256 + n]
                      : wg [(size_t)k * 256 + (n - 256)];
  wT[idx] = (__bf16)v;
}

// ---------- kernel 1: fused kv+gate GEMM, bf16 WMMA, fp32 accumulate ----------
// C(32768 x 512) = A(32768 x 4096) * Wt^T   (Wt stored as [512][4096] bf16)
// Pipeline: A reg-double-buffered (non-temporal global_load_b128),
// B LDS-double-buffered via CDNA5 global_load_async_to_lds_b128 (ASYNCcnt).
__global__ __launch_bounds__(256) void gemm_kvgate(
    const float* __restrict__ hidden,     // (32768, 4096) fp32
    const __bf16* __restrict__ wT,        // (512, 4096)   bf16
    float* __restrict__ kvgate)           // (32768, 512)  fp32
{
  __shared__ alignas(16) __bf16 lA[TILE_M * ASTRIDE];
  __shared__ alignas(16) __bf16 lB[2 * TILE_N * BSTRIDE];

  const int tid  = threadIdx.x;
  const int lane = tid & 31;
  const int wave = tid >> 5;            // 8 waves
  const int wm   = wave & 3;            // 4 strips of 32 rows
  const int wn   = wave >> 2;           // 2 strips of 64 cols
  const int m_base = blockIdx.y * TILE_M;
  const int n_base = blockIdx.x * TILE_N;
  const int l16 = lane & 15;
  const int hiHalf = (lane >> 4);       // 0: lanes 0-15, 1: lanes 16-31

  // A staging: thread owns row (tid>>1), 32-float slice ((tid&1)*32)
  const int arow = tid >> 1;
  const int acol = (tid & 1) * 32;
  const float* aptr = hidden + (size_t)(m_base + arow) * H_DIM + acol;

  v8f acc[2][4];
#pragma unroll
  for (int mt = 0; mt < 2; ++mt)
#pragma unroll
    for (int nt = 0; nt < 4; ++nt)
#pragma unroll
      for (int e = 0; e < 8; ++e) acc[mt][nt][e] = 0.0f;

  f4v areg[8];
  auto loadA = [&](int k0) {
#pragma unroll
    for (int j = 0; j < 8; ++j)
      areg[j] = __builtin_nontemporal_load((const f4v*)(aptr + k0 + j * 4));
  };
  auto storeA = [&]() {
#pragma unroll
    for (int t = 0; t < 4; ++t) {
      uint4 u;
      u.x = pk2bf(areg[2*t][0],   areg[2*t][1]);
      u.y = pk2bf(areg[2*t][2],   areg[2*t][3]);
      u.z = pk2bf(areg[2*t+1][0], areg[2*t+1][1]);
      u.w = pk2bf(areg[2*t+1][2], areg[2*t+1][3]);
      *(uint4*)&lA[arow * ASTRIDE + acol + t * 8] = u;
    }
  };
  // B async staging: 1024 16B-chunks per tile, 4 per thread
  auto asyncB = [&](int k0, int buf) {
#pragma unroll
    for (int j = 0; j < 4; ++j) {
      int c   = tid + j * 256;
      int row = c >> 3;
      int ch  = c & 7;
      uint32_t ldsa = (uint32_t)(uintptr_t)
          &lB[buf * TILE_N * BSTRIDE + row * BSTRIDE + ch * 8];
      const __bf16* gp = wT + (size_t)(n_base + row) * H_DIM + k0 + ch * 8;
      asm volatile("global_load_async_to_lds_b128 %0, %1, off"
                   :: "v"(ldsa), "v"(gp) : "memory");
    }
  };

  // ---- prologue ----
  loadA(0);
  asyncB(0, 0);
  int buf = 0;

  for (int k0 = 0; k0 < H_DIM; k0 += KSTEP) {
    storeA();                                   // consume areg (tile k0)
    const bool more = (k0 + KSTEP < H_DIM);
    if (more) loadA(k0 + KSTEP);                // A loads overlap compute
    asm volatile("s_wait_asynccnt 0" ::: "memory");   // B tile k0 in LDS
    __syncthreads();
    if (more) asyncB(k0 + KSTEP, buf ^ 1);      // B loads overlap compute

    const __bf16* lBb = &lB[buf * TILE_N * BSTRIDE];
#pragma unroll
    for (int s = 0; s < 2; ++s) {               // two k=32 slices
      // A 16x32 frag: lanes 0-15 row=l16 K{0..7,16..23}; 16-31 K{8..15,24..31}
      v16bf afr[2];
#pragma unroll
      for (int mt = 0; mt < 2; ++mt) {
        const __bf16* rp = &lA[(wm * 32 + mt * 16 + l16) * ASTRIDE + s * 32];
        v8bf lo = *(const v8bf*)(rp + hiHalf * 8);
        v8bf hi = *(const v8bf*)(rp + hiHalf * 8 + 16);
#pragma unroll
        for (int e = 0; e < 8; ++e) { afr[mt][e] = lo[e]; afr[mt][e + 8] = hi[e]; }
      }
      // B 32x16 frag: lanes 0-15 col=l16 K{0..7,16..23}; 16-31 K{8..15,24..31}
      v16bf bfr[4];
#pragma unroll
      for (int nt = 0; nt < 4; ++nt) {
        const __bf16* rp = &lBb[(wn * 64 + nt * 16 + l16) * BSTRIDE + s * 32 + hiHalf * 8];
        v8bf lo = *(const v8bf*)(rp);
        v8bf hi = *(const v8bf*)(rp + 16);
#pragma unroll
        for (int e = 0; e < 8; ++e) { bfr[nt][e] = lo[e]; bfr[nt][e + 8] = hi[e]; }
      }
#pragma unroll
      for (int mt = 0; mt < 2; ++mt)
#pragma unroll
        for (int nt = 0; nt < 4; ++nt)
          acc[mt][nt] = __builtin_amdgcn_wmma_f32_16x16x32_bf16(
              false, afr[mt], false, bfr[nt], (short)0, acc[mt][nt], false, false);
    }
    __syncthreads();                            // lA reuse fence
    buf ^= 1;
  }

  // ---- store C: VGPR r -> M = r + (lane<16 ? 0 : 8), N = lane%16 ----
#pragma unroll
  for (int mt = 0; mt < 2; ++mt)
#pragma unroll
    for (int nt = 0; nt < 4; ++nt) {
      int gm = m_base + wm * 32 + mt * 16 + hiHalf * 8;
      int gn = n_base + wn * 64 + nt * 16 + l16;
      float* outp = kvgate + (size_t)gm * N_OUT + gn;
#pragma unroll
      for (int r = 0; r < 8; ++r) outp[(size_t)r * N_OUT] = acc[mt][nt][r];
    }
}

// ---------- kernel 2: windowed online-softmax pool + RoPE ----------
__global__ __launch_bounds__(128) void pool_rope(
    const float* __restrict__ kvgate,   // (32768, 512)
    const float* __restrict__ pb,       // (64, 128)
    float* __restrict__ out)            // (4, 256, 128)
{
  const int d = threadIdx.x;            // feature 0..127
  const int i = blockIdx.x;             // chunk 0..255
  const int b = blockIdx.y;             // batch 0..3

  const size_t rowbase  = (size_t)(b * 8192 + i * 32) * N_OUT;
  const size_t prevbase = (size_t)(b * 8192 + (i - 1) * 32) * N_OUT;

  float m = -INFINITY, s = 0.0f, acc = 0.0f;
  auto upd = [&](float g, float kv) {
    float mn = fmaxf(m, g);
    float sc = __expf(m - mn);
    float e  = __expf(g - mn);
    s   = s * sc + e;
    acc = acc * sc + e * kv;
    m = mn;
  };

  if (i == 0) {
#pragma unroll 4
    for (int j = 0; j < 32; ++j) upd(-1.0e9f + pb[j * 128 + d], 0.0f);
  } else {
#pragma unroll 4
    for (int j = 0; j < 32; ++j) {
      float g  = kvgate[prevbase + (size_t)j * N_OUT + 256 + d] + pb[j * 128 + d];
      float kv = kvgate[prevbase + (size_t)j * N_OUT + d];
      upd(g, kv);
    }
  }
#pragma unroll 4
  for (int j = 0; j < 32; ++j) {
    float g  = kvgate[rowbase + (size_t)j * N_OUT + 384 + d] + pb[(32 + j) * 128 + d];
    float kv = kvgate[rowbase + (size_t)j * N_OUT + 128 + d];
    upd(g, kv);
  }

  float pooled  = acc / s;
  float partner = __shfl_xor(pooled, 1, 32);   // interleaved pair (wave32-safe)

  float res = pooled;
  if (d >= 64) {
    int   r   = d - 64;
    float pos = (float)(i * 32);
    float inv = __powf(10000.0f, -(float)(r & ~1) / 64.0f);
    float fr  = pos * inv;
    float c   = __cosf(fr);
    float sn  = __sinf(fr);
    res = (r & 1) ? (pooled * c + partner * sn)
                  : (pooled * c - partner * sn);
  }
  out[((size_t)b * 256 + i) * 128 + d] = res;
}

// ---------- launcher ----------
extern "C" void kernel_launch(void* const* d_in, const int* in_sizes, int n_in,
                              void* d_out, int out_size, void* d_ws, size_t ws_size,
                              hipStream_t stream) {
  const float* hidden = (const float*)d_in[0];   // (4, 8192, 4096)
  const float* wkv    = (const float*)d_in[1];   // (4096, 256)
  const float* wg     = (const float*)d_in[2];   // (4096, 256)
  const float* pb     = (const float*)d_in[3];   // (64, 128)
  float*       out    = (float*)d_out;           // (4, 256, 128)

  __bf16* wT     = (__bf16*)d_ws;                                     // 4 MB
  float*  kvgate = (float*)((char*)d_ws + (size_t)N_OUT * H_DIM * 2); // 67 MB

  prep_weights<<<(N_OUT * H_DIM + 255) / 256, 256, 0, stream>>>(wkv, wg, wT);

  dim3 g1(N_OUT / TILE_N, M_TOT / TILE_M);       // (4, 256)
  gemm_kvgate<<<g1, 256, 0, stream>>>(hidden, wT, kvgate);

  dim3 g2(256, 4);
  pool_rope<<<g2, 128, 0, stream>>>(kvgate, pb, out);
}
